// PointerNetwork_88905823027917
// MI455X (gfx1250) — matrix operand order repaired and checked by reading, not compiled, not run on hardware
//
#include <hip/hip_runtime.h>
#include <hip/hip_bf16.h>

typedef __attribute__((ext_vector_type(16))) _Float16 v16h;
typedef __attribute__((ext_vector_type(8)))  float    v8f;

#define LANE (threadIdx.x & 31)
#define WAVE (threadIdx.x >> 5)

static __device__ __forceinline__ float sigf(float x) {
    return 1.0f / (1.0f + __expf(-x));
}

// B-fragment for V_WMMA_F32_16X16X32_F16: B is W stored row-major (N x K),
// WMMA wants B[k][n] = W[n][k]. Lane n = l&15; b[j] = W[n][kb+8g+j], b[8+j] = W[n][kb+16+8g+j].
static __device__ __forceinline__ v16h load_b16(const _Float16* W, int nrow, int K, int kb) {
    const int g = (LANE >> 4) & 1;
    const _Float16* p = W + nrow * K + kb + 8 * g;
    v16h b;
#pragma unroll
    for (int j = 0; j < 8; ++j) { b[j] = p[j]; b[8 + j] = p[16 + j]; }
    return b;
}

// A-fragment: A row-major (M x K). Lane m = l&15; a[j] = A[m][kb+8g+j], a[8+j] = A[m][kb+16+8g+j].
static __device__ __forceinline__ v16h load_a16(const _Float16* A, int row, int K, int kb) {
    const int g = (LANE >> 4) & 1;
    const _Float16* p = A + row * K + kb + 8 * g;
    v16h a;
#pragma unroll
    for (int j = 0; j < 8; ++j) { a[j] = p[j]; a[8 + j] = p[16 + j]; }
    return a;
}

// ---------------- small elementwise kernels ----------------

// enc_in = inputs @ W_enc.T + b_enc ; inputs (B*L, 2), W_enc (256, 2)
__global__ void enc_linear_kernel(const float* __restrict__ inputs,
                                  const float* __restrict__ W_enc,
                                  const float* __restrict__ b_enc,
                                  float* __restrict__ enc32,
                                  _Float16* __restrict__ enc16) {
    int e  = blockIdx.x * 256 + threadIdx.x;   // 4096*256 total
    int bl = e >> 8;
    int hh = e & 255;
    float v = inputs[bl * 2] * W_enc[hh * 2] + inputs[bl * 2 + 1] * W_enc[hh * 2 + 1] + b_enc[hh];
    enc32[e] = v;
    enc16[e] = (_Float16)v;
}

__global__ void cvt_f16_kernel(const float* __restrict__ in, _Float16* __restrict__ out, int n) {
    int i = blockIdx.x * 256 + threadIdx.x;
    if (i < n) out[i] = (_Float16)in[i];
}

// dec_in[b,t,:] = enc_in[b, targets[b,(t-1) mod L], :]
__global__ void gather_dec_kernel(const _Float16* __restrict__ enc16,
                                  const int* __restrict__ targets,
                                  _Float16* __restrict__ dec16) {
    int e  = blockIdx.x * 256 + threadIdx.x;   // 4096*256
    int hh = e & 255;
    int bl = e >> 8;
    int b  = bl >> 7;
    int t  = bl & 127;
    int src = targets[b * 128 + ((t + 127) & 127)];
    dec16[e] = enc16[(b * 128 + src) * 256 + hh];
}

// inv[b, pos] = t  such that targets[b,t] == pos  (targets is a permutation)
__global__ void inv_perm_kernel(const int* __restrict__ targets, int* __restrict__ inv) {
    int idx = blockIdx.x * 256 + threadIdx.x;  // 4096
    int b = idx >> 7, t = idx & 127;
    inv[b * 128 + targets[idx]] = t;
}

// kT[b, a, j] = k[(b*128+j)*256 + a]
__global__ void transpose_k_kernel(const float* __restrict__ k, float* __restrict__ kT) {
    int e  = blockIdx.x * 256 + threadIdx.x;   // 4096*256
    int bl = e >> 8;
    int a  = e & 255;
    int b = bl >> 7, j = bl & 127;
    kT[(b * 256 + a) * 128 + j] = k[e];
}

// ---------------- generic f16 WMMA GEMM: C(MxN) = A(MxK) * W(NxK)^T + bias ----------------
__global__ void gemm_f16_wmma_kernel(const _Float16* __restrict__ A,
                                     const _Float16* __restrict__ W,
                                     const float* __restrict__ bias,
                                     float* __restrict__ C,
                                     int M, int N, int K) {
    int tile = blockIdx.x * 4 + WAVE;
    int ntm  = M >> 4;
    int mi = tile % ntm;
    int ni = tile / ntm;
    int lm = LANE & 15, g = LANE >> 4;

    v8f acc = {};
    for (int kb = 0; kb < K; kb += 32) {
        v16h a = load_a16(A, mi * 16 + lm, K, kb);
        v16h b = load_b16(W, ni * 16 + lm, K, kb);
        acc = __builtin_amdgcn_wmma_f32_16x16x32_f16(false, a, false, b, (short)0, acc, false, false);
    }
    int col = ni * 16 + lm;
    float bv = bias ? bias[col] : 0.0f;
#pragma unroll
    for (int v = 0; v < 8; ++v) {
        int row = mi * 16 + v + 8 * g;
        C[row * N + col] = acc[v] + bv;
    }
}

// ---------------- persistent single-workgroup GRU (128 steps, h in LDS) ----------------
// GI: (B*L, 768) precomputed x@Wih.T+bih (row = b*128+t). Whh16: (768,256) f16.
// Wave w: m-block = w&1, n-block hh0 = (w>>1)*16; owns r/z/n tiles for those rows/cols,
// and performs the gate fusion in registers directly from the WMMA C layout.
__global__ __launch_bounds__(1024) void gru_wmma_kernel(const float* __restrict__ GI,
                                                        const _Float16* __restrict__ Whh16,
                                                        const float* __restrict__ bhh,
                                                        const float* __restrict__ h_init,
                                                        float* __restrict__ h_out,
                                                        _Float16* __restrict__ out16) {
    __shared__ float    h32[32 * 256];
    __shared__ _Float16 h16[32 * 256];

    for (int i = threadIdx.x; i < 32 * 256; i += 1024) {
        float v = h_init ? h_init[i] : 0.0f;
        h32[i] = v;
        h16[i] = (_Float16)v;
    }
    __syncthreads();

    const int w   = WAVE;
    const int m0  = (w & 1) * 16;
    const int hh0 = (w >> 1) * 16;
    const int lm  = LANE & 15, g = LANE >> 4;
    const int col = hh0 + lm;

    for (int t = 0; t < 128; ++t) {
        v8f aR = {}, aZ = {}, aN = {};
        for (int kb = 0; kb < 256; kb += 32) {
            v16h af = load_a16(h16, m0 + lm, 256, kb);
            v16h bR = load_b16(Whh16, col, 256, kb);          // n = col        (reset gate rows)
            v16h bZ = load_b16(Whh16, 256 + col, 256, kb);    // n = 256 + col  (update gate rows)
            v16h bN = load_b16(Whh16, 512 + col, 256, kb);    // n = 512 + col  (new gate rows)
            aR = __builtin_amdgcn_wmma_f32_16x16x32_f16(false, af, false, bR, (short)0, aR, false, false);
            aZ = __builtin_amdgcn_wmma_f32_16x16x32_f16(false, af, false, bZ, (short)0, aZ, false, false);
            aN = __builtin_amdgcn_wmma_f32_16x16x32_f16(false, af, false, bN, (short)0, aN, false, false);
        }
        __syncthreads();   // all reads of old h16 done
#pragma unroll
        for (int v = 0; v < 8; ++v) {
            int m = m0 + v + 8 * g;
            const float* gi = GI + (m * 128 + t) * 768;
            float r  = sigf(gi[col]        + aR[v] + bhh[col]);
            float z  = sigf(gi[256 + col]  + aZ[v] + bhh[256 + col]);
            float nn = tanhf(gi[512 + col] + r * (aN[v] + bhh[512 + col]));
            float hn = (1.0f - z) * nn + z * h32[m * 256 + col];
            h32[m * 256 + col] = hn;
            h16[m * 256 + col] = (_Float16)hn;
            out16[(m * 128 + t) * 256 + col] = (_Float16)hn;
        }
        __syncthreads();   // new h visible before next step's reads
    }
    if (h_out)
        for (int i = threadIdx.x; i < 32 * 256; i += 1024) h_out[i] = h32[i];
}

// ---------------- additive attention + permutation mask ----------------
// out[b,i,j] = mask ? sum_a v[a]*tanh(q[b,i,a]+k[b,j,a]) : -1e9
// 4 query rows per block: each k element loaded once feeds 4 independent
// tanh+fma chains (ILP to hide v_tanh latency, 1/4 the L2 traffic).
__global__ void attn_kernel(const float* __restrict__ q,
                            const float* __restrict__ kT,   // (B, 256, 128)
                            const float* __restrict__ vatt,
                            const int* __restrict__ inv,
                            float* __restrict__ out) {
    int blk = blockIdx.x;              // 1024 blocks: b = blk>>5, i0 = (blk&31)*4
    int b  = blk >> 5;
    int i0 = (blk & 31) * 4;
    __shared__ float qs[4 * 256], vs[256];
    int tid = threadIdx.x;             // 128 threads, j = tid
#pragma unroll
    for (int r = 0; r < 4; ++r) {
        qs[r * 256 + tid]       = q[(b * 128 + i0 + r) * 256 + tid];
        qs[r * 256 + tid + 128] = q[(b * 128 + i0 + r) * 256 + tid + 128];
    }
    vs[tid]       = vatt[tid];
    vs[tid + 128] = vatt[tid + 128];
    __syncthreads();

    const float* kp = kT + b * 256 * 128 + tid;
    float acc0 = 0.0f, acc1 = 0.0f, acc2 = 0.0f, acc3 = 0.0f;
#pragma unroll 2
    for (int a = 0; a < 256; ++a) {
        float kv = kp[a * 128];
        float vv = vs[a];
        acc0 += vv * tanhf(qs[a]       + kv);
        acc1 += vv * tanhf(qs[256 + a] + kv);
        acc2 += vv * tanhf(qs[512 + a] + kv);
        acc3 += vv * tanhf(qs[768 + a] + kv);
    }

    int iv = inv[b * 128 + tid];
    float o0 = (iv >= i0 + 0) ? acc0 : -1e9f;
    float o1 = (iv >= i0 + 1) ? acc1 : -1e9f;
    float o2 = (iv >= i0 + 2) ? acc2 : -1e9f;
    float o3 = (iv >= i0 + 3) ? acc3 : -1e9f;
    out[(b * 128 + i0 + 0) * 128 + tid] = o0;
    out[(b * 128 + i0 + 1) * 128 + tid] = o1;
    out[(b * 128 + i0 + 2) * 128 + tid] = o2;
    out[(b * 128 + i0 + 3) * 128 + tid] = o3;
}

// ---------------- launcher ----------------
extern "C" void kernel_launch(void* const* d_in, const int* in_sizes, int n_in,
                              void* d_out, int out_size, void* d_ws, size_t ws_size,
                              hipStream_t stream) {
    const float* inputs  = (const float*)d_in[0];
    const int*   targets = (const int*)d_in[1];
    const float* W_enc   = (const float*)d_in[2];
    const float* b_enc   = (const float*)d_in[3];
    const float* Wih_e   = (const float*)d_in[4];
    const float* Whh_e   = (const float*)d_in[5];
    const float* bih_e   = (const float*)d_in[6];
    const float* bhh_e   = (const float*)d_in[7];
    const float* Wih_d   = (const float*)d_in[8];
    const float* Whh_d   = (const float*)d_in[9];
    const float* bih_d   = (const float*)d_in[10];
    const float* bhh_d   = (const float*)d_in[11];
    const float* Wq      = (const float*)d_in[12];
    const float* Wk      = (const float*)d_in[13];
    const float* v_att   = (const float*)d_in[14];

    const int BL = 32 * 128;       // 4096
    char* ws = (char*)d_ws;
    size_t off = 0;
    auto alloc = [&](size_t bytes) { char* p = ws + off; off += (bytes + 255) & ~size_t(255); return p; };

    float*    enc32    = (float*)   alloc(BL * 256 * 4);
    _Float16* enc16    = (_Float16*)alloc(BL * 256 * 2);
    _Float16* dec16    = (_Float16*)alloc(BL * 256 * 2);
    _Float16* Wihe16   = (_Float16*)alloc(768 * 256 * 2);
    _Float16* Whhe16   = (_Float16*)alloc(768 * 256 * 2);
    _Float16* Wihd16   = (_Float16*)alloc(768 * 256 * 2);
    _Float16* Whhd16   = (_Float16*)alloc(768 * 256 * 2);
    _Float16* Wq16     = (_Float16*)alloc(256 * 256 * 2);
    _Float16* Wk16     = (_Float16*)alloc(256 * 256 * 2);
    float*    GIe      = (float*)   alloc((size_t)BL * 768 * 4);
    float*    GId      = (float*)   alloc((size_t)BL * 768 * 4);
    _Float16* encout16 = (_Float16*)alloc(BL * 256 * 2);
    _Float16* decout16 = (_Float16*)alloc(BL * 256 * 2);
    float*    henc     = (float*)   alloc(32 * 256 * 4);
    float*    q32      = (float*)   alloc(BL * 256 * 4);
    float*    k32      = (float*)   alloc(BL * 256 * 4);
    float*    kT       = (float*)   alloc(BL * 256 * 4);
    int*      invp     = (int*)     alloc(BL * 4);
    (void)ws_size; (void)in_sizes; (void)n_in; (void)out_size;

    // 1. encoder linear (f32 + f16)
    enc_linear_kernel<<<BL, 256, 0, stream>>>(inputs, W_enc, b_enc, enc32, enc16);
    // 2. weight conversions
    cvt_f16_kernel<<<(768 * 256 + 255) / 256, 256, 0, stream>>>(Wih_e, Wihe16, 768 * 256);
    cvt_f16_kernel<<<(768 * 256 + 255) / 256, 256, 0, stream>>>(Whh_e, Whhe16, 768 * 256);
    cvt_f16_kernel<<<(768 * 256 + 255) / 256, 256, 0, stream>>>(Wih_d, Wihd16, 768 * 256);
    cvt_f16_kernel<<<(768 * 256 + 255) / 256, 256, 0, stream>>>(Whh_d, Whhd16, 768 * 256);
    cvt_f16_kernel<<<(256 * 256 + 255) / 256, 256, 0, stream>>>(Wq, Wq16, 256 * 256);
    cvt_f16_kernel<<<(256 * 256 + 255) / 256, 256, 0, stream>>>(Wk, Wk16, 256 * 256);
    // 3. permutation inverse + decoder-input gather
    inv_perm_kernel<<<BL / 256, 256, 0, stream>>>(targets, invp);
    gather_dec_kernel<<<BL, 256, 0, stream>>>(enc16, targets, dec16);
    // 4. hoisted input-gate GEMMs: (4096x768) = (4096x256)*(768x256)^T + bih
    gemm_f16_wmma_kernel<<<(256 * 48) / 4, 128, 0, stream>>>(enc16, Wihe16, bih_e, GIe, BL, 768, 256);
    gemm_f16_wmma_kernel<<<(256 * 48) / 4, 128, 0, stream>>>(dec16, Wihd16, bih_d, GId, BL, 768, 256);
    // 5. sequential GRUs (single persistent workgroup each)
    gru_wmma_kernel<<<1, 1024, 0, stream>>>(GIe, Whhe16, bhh_e, nullptr, henc, encout16);
    gru_wmma_kernel<<<1, 1024, 0, stream>>>(GId, Whhd16, bhh_d, henc, nullptr, decout16);
    // 6. attention projections
    gemm_f16_wmma_kernel<<<(256 * 16) / 4, 128, 0, stream>>>(decout16, Wq16, nullptr, q32, BL, 256, 256);
    gemm_f16_wmma_kernel<<<(256 * 16) / 4, 128, 0, stream>>>(encout16, Wk16, nullptr, k32, BL, 256, 256);
    // 7. k transpose for coalesced attention loads
    transpose_k_kernel<<<BL, 256, 0, stream>>>(k32, kT);
    // 8. additive attention + mask (4 query rows per block)
    attn_kernel<<<BL / 4, 128, 0, stream>>>(q32, kT, v_att, invp, (float*)d_out);
}
